// Predictor_30786325578115
// MI455X (gfx1250) — matrix operand order repaired and checked by reading, not compiled, not run on hardware
//
#include <hip/hip_runtime.h>

#define D_DIM    1024
#define VOCAB_SZ 1024
#define BATCH    32
#define TSEQ     513            // U+1
#define USEQ     512
#define N3       (3 * D_DIM)    // 3072
#define MTOT     (BATCH * TSEQ) // 16416
#define MT48     (MTOT / 48)    // 342 M-tiles of 48 rows
#define GRU_WGS  64
#define CSTR     1032           // LDS column stride in bf16 elems (1024 + 16B pad)

// Feature-detect the gfx1250 async global->LDS path (compile-safe probe).
#if defined(__has_builtin)
#  if __has_builtin(__builtin_amdgcn_global_load_async_to_lds_b128) && \
      __has_builtin(__builtin_amdgcn_s_wait_asynccnt)
#    define HAVE_ASYNC_LDS 1
#  endif
#endif
#ifndef HAVE_ASYNC_LDS
#  define HAVE_ASYNC_LDS 0
#endif

typedef __attribute__((ext_vector_type(16))) __bf16 v16bf;
typedef __attribute__((ext_vector_type(8)))  __bf16 v8bf;
typedef __attribute__((ext_vector_type(8)))  float  v8f;

#if HAVE_ASYNC_LDS
// Exact pointee type from the compiler diagnostic: int __attribute__((vector_size(16)))
typedef int v4i_vs __attribute__((vector_size(16)));
typedef __attribute__((address_space(1))) v4i_vs g_v4i;  // global (prints as __device__)
typedef __attribute__((address_space(3))) v4i_vs l_v4i;  // LDS
#endif

__device__ __forceinline__ v8f vzero8() {
  v8f z = {0.f, 0.f, 0.f, 0.f, 0.f, 0.f, 0.f, 0.f};
  return z;
}

// Fragment loader: two contiguous 8-element bf16 runs (16B each).
__device__ __forceinline__ v16bf load_frag_bf16(const __bf16* p0, const __bf16* p1) {
  v8bf lo = *(const v8bf*)p0;
  v8bf hi = *(const v8bf*)p1;
  v16bf f;
#pragma unroll
  for (int i = 0; i < 8; ++i) { f[i] = lo[i]; f[i + 8] = hi[i]; }
  return f;
}

__device__ __forceinline__ v8f wmma_bf16(v16bf a, v16bf b, v8f c) {
  return __builtin_amdgcn_wmma_f32_16x16x32_bf16(
      /*neg_a=*/false, a, /*neg_b=*/false, b,
      /*c_mod=*/(short)0, c, /*reuse_a=*/false, /*reuse_b=*/false);
}

__device__ __forceinline__ float sigmoidf_(float x) {
  return 1.f / (1.f + __expf(-x));
}

// ---------------------------------------------------------------- prep kernels

__global__ void k_cast_bf16(const float* __restrict__ in, __bf16* __restrict__ out, int n) {
  int i = blockIdx.x * 256 + threadIdx.x;
  if (i < n) out[i] = (__bf16)in[i];
}

// in: fp32 [R][C] row-major  ->  out: bf16 [C][R] (transposed, K-contiguous rows)
__global__ void k_transpose_bf16(const float* __restrict__ in, __bf16* __restrict__ out,
                                 int R, int C) {
  int i = blockIdx.x * 256 + threadIdx.x;
  if (i < R * C) {
    int r = i / C, c = i % C;
    out[(size_t)c * R + r] = (__bf16)in[i];
  }
}

// ---------------------------------------------------------------- input GEMM
// xp[r][n] = embed_bf16[token(r)] @ Wx + bx ;  r = b*513 + t
// Wave tile 48x64 (3 M-frags x 4 N-frags): 14 x 16B loads per 12 WMMAs.
// All 7 fragments are loaded into distinct registers BEFORE the WMMA block so
// the 12 matrix ops issue without per-subtile loadcnt waits; K unrolled x2 so
// the next iteration's loads overlap the current WMMA chain.
__global__ __launch_bounds__(256, 2) void k_gemm_in(
    const int* __restrict__ y, const __bf16* __restrict__ E,
    const __bf16* __restrict__ WxT, const float* __restrict__ bx,
    float* __restrict__ xp) {
  const int Nt = N3 / 64;  // 48
  int gw   = (blockIdx.x * 256 + (int)threadIdx.x) >> 5;
  int lane = threadIdx.x & 31;
  int tm = gw % MT48, tn = gw / MT48;
  if (tn >= Nt) return;
  int lm = lane & 15, hi = lane >> 4;
  int m0 = tm * 48, n0 = tn * 64;
  int base0 = hi * 8, kb = hi * 8;

  const __bf16* arow[3];
#pragma unroll
  for (int mi = 0; mi < 3; ++mi) {
    int r = m0 + mi * 16 + lm;
    int b = r / TSEQ, t = r % TSEQ;
    int tok = (t == 0) ? 0 : y[b * USEQ + t - 1];  // NULL_INDEX = 0
    arow[mi] = E + (size_t)tok * D_DIM;
  }
  const __bf16* brow[4];
#pragma unroll
  for (int nj = 0; nj < 4; ++nj)
    brow[nj] = WxT + (size_t)(n0 + nj * 16 + lm) * D_DIM;

  v8f acc[3][4];
#pragma unroll
  for (int mi = 0; mi < 3; ++mi)
#pragma unroll
    for (int nj = 0; nj < 4; ++nj) acc[mi][nj] = vzero8();

#pragma unroll 2
  for (int k0 = 0; k0 < D_DIM; k0 += 32) {
    v16bf a[3], b[4];
#pragma unroll
    for (int mi = 0; mi < 3; ++mi)
      a[mi] = load_frag_bf16(arow[mi] + k0 + base0, arow[mi] + k0 + 16 + base0);
#pragma unroll
    for (int nj = 0; nj < 4; ++nj)
      b[nj] = load_frag_bf16(brow[nj] + k0 + kb, brow[nj] + k0 + kb + 16);
#pragma unroll
    for (int nj = 0; nj < 4; ++nj) {
      acc[0][nj] = wmma_bf16(a[0], b[nj], acc[0][nj]);
      acc[1][nj] = wmma_bf16(a[1], b[nj], acc[1][nj]);
      acc[2][nj] = wmma_bf16(a[2], b[nj], acc[2][nj]);
    }
    if (k0 + 32 < D_DIM) {
      __builtin_prefetch(arow[0] + k0 + 32, 0, 1);
      __builtin_prefetch(brow[0] + k0 + 32, 0, 1);
    }
  }

#pragma unroll
  for (int nj = 0; nj < 4; ++nj) {
    int col = n0 + nj * 16 + lm;
    float bb = bx[col];
#pragma unroll
    for (int mi = 0; mi < 3; ++mi)
#pragma unroll
      for (int i = 0; i < 8; ++i) {
        int row = m0 + mi * 16 + i + hi * 8;
        xp[(size_t)row * N3 + col] = acc[mi][nj][i] + bb;
      }
  }
}

// ---------------------------------------------------------------- grid barrier

__device__ __forceinline__ void grid_sync(unsigned* cnt, unsigned target) {
  __syncthreads();
  if (threadIdx.x == 0) {
    __threadfence();          // release my writes
    atomicAdd(cnt, 1u);
    while (__hip_atomic_load(cnt, __ATOMIC_ACQUIRE, __HIP_MEMORY_SCOPE_AGENT) < target) {
      __builtin_amdgcn_s_sleep(1);
    }
  }
  __syncthreads();
}

// ---------------------------------------------------------------- GRU recurrence
// 64 persistent workgroups x 64 threads (2 waves). Workgroup wg owns d-slice
// [wg*16, wg*16+16): its 48 Wh columns (z/r/h gates) are staged ONCE in 96KB
// of LDS (CDNA5: 320KB/WGP) before the 513-step loop via the gfx1250 async
// global->LDS path (ASYNCcnt-tracked). Per-step B fragments are ds_load_b128
// pairs (conflict-free via 16B column padding). Wave w handles batch rows
// [w*16, w*16+16). h state: bf16 in global (WMMA A operand) + exact fp32
// per-lane registers for the elementwise update.
__global__ __launch_bounds__(64, 1) void k_gru(
    const float* __restrict__ xp, const __bf16* __restrict__ WhT,
    const float* __restrict__ bh, const float* __restrict__ h0,
    __bf16* __restrict__ hbuf,   // [2][BATCH][D] bf16, double-buffered
    __bf16* __restrict__ hsb,    // [BATCH*TSEQ][D] bf16
    unsigned* __restrict__ cnt) {
  extern __shared__ __bf16 ldsWh[];  // 48 * CSTR bf16 = 99072 B
  const int tid = threadIdx.x;
  const int d0  = blockIdx.x * 16;

  // Stage this block's 48 Wh columns (K-contiguous rows of WhT) into LDS.
#if HAVE_ASYNC_LDS
  for (int idx = tid; idx < 48 * (D_DIM / 8); idx += 64) {
    int c = idx >> 7, chunk = idx & 127;     // 128 x 8-elem (16B) chunks / column
    int g = c >> 4, n = c & 15;
    const __bf16* src = WhT + (size_t)(g * D_DIM + d0 + n) * D_DIM + chunk * 8;
    __bf16* dst = ldsWh + c * CSTR + chunk * 8;
    __builtin_amdgcn_global_load_async_to_lds_b128((g_v4i*)src, (l_v4i*)dst, 0, 0);
  }
  __builtin_amdgcn_s_wait_asynccnt(0);
#else
  for (int idx = tid; idx < 48 * (D_DIM / 8); idx += 64) {
    int c = idx >> 7, chunk = idx & 127;     // 128 x 8-elem chunks per column
    int g = c >> 4, n = c & 15;
    const v8bf* src = (const v8bf*)(WhT + (size_t)(g * D_DIM + d0 + n) * D_DIM + chunk * 8);
    *(v8bf*)(ldsWh + c * CSTR + chunk * 8) = *src;
  }
#endif

  // Broadcast h0 into buffer 0 for this block's d-slice (all batch rows).
  for (int idx = tid; idx < BATCH * 16; idx += 64) {
    int b = idx >> 4, j = idx & 15;
    hbuf[(size_t)b * D_DIM + d0 + j] = (__bf16)h0[d0 + j];
  }
  __syncthreads();
  unsigned epoch = 1;
  grid_sync(cnt, epoch * GRU_WGS);

  const int lane = tid & 31, wave = tid >> 5;
  const int lm = lane & 15, hi = lane >> 4;
  const int m0 = wave * 16;
  const int d  = d0 + lm;
  const int base0 = hi * 8, kb = hi * 8;

  const float bz  = bh[d];
  const float br  = bh[D_DIM + d];
  const float bhh = bh[2 * D_DIM + d];
  const __bf16* bcol0 = ldsWh + (size_t)(0 * 16 + lm) * CSTR;
  const __bf16* bcol1 = ldsWh + (size_t)(1 * 16 + lm) * CSTR;
  const __bf16* bcol2 = ldsWh + (size_t)(2 * 16 + lm) * CSTR;

  // h_prev for this lane's 8 (b,d) C-fragment slots stays in fp32 registers:
  // the C layout (m = i + hi*8, n = lm) is step-invariant.
  float hreg[8];
#pragma unroll
  for (int i = 0; i < 8; ++i) hreg[i] = h0[d];

  for (int t = 0; t < TSEQ; ++t) {
    const __bf16* hc = hbuf + (size_t)(t & 1) * BATCH * D_DIM;
    __bf16*       hn = hbuf + (size_t)((t + 1) & 1) * BATCH * D_DIM;

    // Prefetch next step's xp slice (one 64B line per row per gate).
    if (t + 1 < TSEQ && lane < 16) {
      const float* p = xp + (size_t)((m0 + lane) * TSEQ + t + 1) * N3 + d0;
      __builtin_prefetch(p, 0, 1);
      __builtin_prefetch(p + D_DIM, 0, 1);
      __builtin_prefetch(p + 2 * D_DIM, 0, 1);
    }

    v8f az = vzero8(), ar = vzero8(), ah = vzero8();
    const __bf16* arow = hc + (size_t)(m0 + lm) * D_DIM;
#pragma unroll 4
    for (int k0 = 0; k0 < D_DIM; k0 += 32) {
      v16bf af = load_frag_bf16(arow + k0 + base0, arow + k0 + 16 + base0);
      v16bf b0 = load_frag_bf16(bcol0 + k0 + kb, bcol0 + k0 + kb + 16);
      v16bf b1 = load_frag_bf16(bcol1 + k0 + kb, bcol1 + k0 + kb + 16);
      v16bf b2 = load_frag_bf16(bcol2 + k0 + kb, bcol2 + k0 + kb + 16);
      az = wmma_bf16(af, b0, az);
      ar = wmma_bf16(af, b1, ar);
      ah = wmma_bf16(af, b2, ah);
    }

#pragma unroll
    for (int i = 0; i < 8; ++i) {
      int b = m0 + i + hi * 8;
      const float* xpr = xp + (size_t)(b * TSEQ + t) * N3 + d;
      float z    = sigmoidf_(xpr[0]         + az[i] + bz);
      float r    = sigmoidf_(xpr[D_DIM]     + ar[i] + br);
      float hcan = tanhf    (xpr[2 * D_DIM] + r * (ah[i] + bhh));  // reset_after
      float hnew = z * hreg[i] + (1.f - z) * hcan;
      hreg[i] = hnew;
      __bf16 hb = (__bf16)hnew;
      hn[(size_t)b * D_DIM + d] = hb;
      hsb[(size_t)(b * TSEQ + t) * D_DIM + d] = hb;
    }

    ++epoch;
    grid_sync(cnt, epoch * GRU_WGS);
  }
}

// ---------------------------------------------------------------- output GEMM
// out[r][n] = hs[r] @ Wd + bd   (A rows are bf16 straight from the recurrence)
__global__ __launch_bounds__(256, 2) void k_gemm_out(
    const __bf16* __restrict__ hsb, const __bf16* __restrict__ WdT,
    const float* __restrict__ bd, float* __restrict__ out) {
  const int Nn = D_DIM;
  const int Nt = Nn / 64;  // 16
  int gw   = (blockIdx.x * 256 + (int)threadIdx.x) >> 5;
  int lane = threadIdx.x & 31;
  int tm = gw % MT48, tn = gw / MT48;
  if (tn >= Nt) return;
  int lm = lane & 15, hi = lane >> 4;
  int m0 = tm * 48, n0 = tn * 64;
  int base0 = hi * 8, kb = hi * 8;

  const __bf16* arow[3];
#pragma unroll
  for (int mi = 0; mi < 3; ++mi)
    arow[mi] = hsb + (size_t)(m0 + mi * 16 + lm) * D_DIM;
  const __bf16* brow[4];
#pragma unroll
  for (int nj = 0; nj < 4; ++nj)
    brow[nj] = WdT + (size_t)(n0 + nj * 16 + lm) * D_DIM;

  v8f acc[3][4];
#pragma unroll
  for (int mi = 0; mi < 3; ++mi)
#pragma unroll
    for (int nj = 0; nj < 4; ++nj) acc[mi][nj] = vzero8();

#pragma unroll 2
  for (int k0 = 0; k0 < D_DIM; k0 += 32) {
    v16bf a[3], b[4];
#pragma unroll
    for (int mi = 0; mi < 3; ++mi)
      a[mi] = load_frag_bf16(arow[mi] + k0 + base0, arow[mi] + k0 + 16 + base0);
#pragma unroll
    for (int nj = 0; nj < 4; ++nj)
      b[nj] = load_frag_bf16(brow[nj] + k0 + kb, brow[nj] + k0 + kb + 16);
#pragma unroll
    for (int nj = 0; nj < 4; ++nj) {
      acc[0][nj] = wmma_bf16(a[0], b[nj], acc[0][nj]);
      acc[1][nj] = wmma_bf16(a[1], b[nj], acc[1][nj]);
      acc[2][nj] = wmma_bf16(a[2], b[nj], acc[2][nj]);
    }
    if (k0 + 32 < D_DIM) {
      __builtin_prefetch(arow[0] + k0 + 32, 0, 1);
      __builtin_prefetch(brow[0] + k0 + 32, 0, 1);
    }
  }

#pragma unroll
  for (int nj = 0; nj < 4; ++nj) {
    int col = n0 + nj * 16 + lm;
    float bb = bd[col];
#pragma unroll
    for (int mi = 0; mi < 3; ++mi)
#pragma unroll
      for (int i = 0; i < 8; ++i) {
        int row = m0 + mi * 16 + i + hi * 8;
        out[(size_t)row * Nn + col] = acc[mi][nj][i] + bb;
      }
  }
}

// ---------------------------------------------------------------- launch

extern "C" void kernel_launch(void* const* d_in, const int* in_sizes, int n_in,
                              void* d_out, int out_size, void* d_ws, size_t ws_size,
                              hipStream_t stream) {
  (void)in_sizes; (void)n_in; (void)out_size; (void)ws_size;
  const int*   y     = (const int*)  d_in[0];
  const float* embed = (const float*)d_in[1];
  const float* Wx    = (const float*)d_in[2];
  const float* Wh    = (const float*)d_in[3];
  const float* bx    = (const float*)d_in[4];
  const float* bhp   = (const float*)d_in[5];
  const float* Wd    = (const float*)d_in[6];
  const float* bd    = (const float*)d_in[7];
  const float* h0    = (const float*)d_in[8];
  float* out = (float*)d_out;

  char* ws = (char*)d_ws;
  size_t off = 0;
  unsigned* cnt = (unsigned*)(ws + off);           off += 256;
  __bf16* Eb    = (__bf16*)(ws + off);             off += (size_t)VOCAB_SZ * D_DIM * 2;
  __bf16* WxT   = (__bf16*)(ws + off);             off += (size_t)N3 * D_DIM * 2;
  __bf16* WhT   = (__bf16*)(ws + off);             off += (size_t)N3 * D_DIM * 2;
  __bf16* WdT   = (__bf16*)(ws + off);             off += (size_t)D_DIM * D_DIM * 2;
  __bf16* hbuf  = (__bf16*)(ws + off);             off += (size_t)2 * BATCH * D_DIM * 2;
  __bf16* hsb   = (__bf16*)(ws + off);             off += (size_t)MTOT * D_DIM * 2;
  off = (off + 255) & ~(size_t)255;
  float*  xp    = (float*)(ws + off);              off += (size_t)MTOT * N3 * 4;

  (void)hipMemsetAsync(cnt, 0, 256, stream);

  // One-time weight quantization / transposition (bf16, K-contiguous B layout).
  k_cast_bf16<<<(VOCAB_SZ * D_DIM + 255) / 256, 256, 0, stream>>>(embed, Eb, VOCAB_SZ * D_DIM);
  k_transpose_bf16<<<(D_DIM * N3 + 255) / 256, 256, 0, stream>>>(Wx, WxT, D_DIM, N3);
  k_transpose_bf16<<<(D_DIM * N3 + 255) / 256, 256, 0, stream>>>(Wh, WhT, D_DIM, N3);
  k_transpose_bf16<<<(D_DIM * D_DIM + 255) / 256, 256, 0, stream>>>(Wd, WdT, D_DIM, D_DIM);

  // Input projection: M=16416, N=3072 -> 342*48 wave-tiles / 8 waves per block.
  k_gemm_in<<<MT48 * (N3 / 64) / 8, 256, 0, stream>>>(y, Eb, WxT, bx, xp);

  // Persistent GRU recurrence: 64 WGs x 2 waves, 96KB LDS, barrier per step.
  k_gru<<<GRU_WGS, 64, 48 * CSTR * sizeof(__bf16), stream>>>(xp, WhT, bhp, h0, hbuf, hsb, cnt);

  // Output projection: M=16416, N=1024 -> 342*16 wave-tiles / 8 per block.
  k_gemm_out<<<MT48 * (D_DIM / 64) / 8, 256, 0, stream>>>(hsb, WdT, bd, out);
}
